// KpcaStd_30863634989789
// MI455X (gfx1250) — matrix operand order repaired and checked by reading, not compiled, not run on hardware
//
#include <hip/hip_runtime.h>
#include <hip/hip_bf16.h>

// ---------------------------------------------------------------------------
// KPCA-style fused loss for MI455X (gfx1250, wave32, WMMA + async-LDS copies).
//   K[i,j] = exp(-(sq[i]+sq[j]-2*x_i.x_j)/2)   (sigma=1)
//   F = K @ H   (K symmetric => E = H^T K = F^T)
//   loss1 = sum_h invl[h] * sum_i F[i,h]^2  / (-2)
//   loss2 = sum_{i,h} F[i,h]*H[i,h] / 2
//   out = s + 0.05 s^2
// K (256 MB > 192 MB L2) is never materialized; each block fuses a 64-row
// slab of F with double-buffered 32-column j-tiles staged asynchronously.
// ---------------------------------------------------------------------------

typedef unsigned short u16;
typedef unsigned int   u32;
typedef __attribute__((ext_vector_type(16))) __bf16 v16bf;
typedef __attribute__((ext_vector_type(8)))  float  v8f;

#define NROWS 8192
#define DDIM  256
#define HDIM  512
#define BI    64          // F rows per block
#define BJ    32          // K columns per j-iteration (double-buffered)
#define NJ    (NROWS/BJ)  // 256 j-iterations
#define NBLK  (NROWS/BI)  // 128 blocks

// LDS strides (bf16 halves): multiples of 8 -> 16B alignment; padded vs banks.
#define A_LD  264   // sA  [64][256]
#define XT_LD 40    // sXT [2][256][32]
#define H_LD  520   // sHb [2][32][512]
#define K_LD  40    // sKt [64][32]

// ---- optional CDNA5 async global->LDS path (ASYNCcnt), with safe fallback --
#if defined(__has_builtin)
#if __has_builtin(__builtin_amdgcn_global_load_async_to_lds_b128)
#define HAVE_ASYNC_LDS 1
#endif
#endif

// Builtin expects: (v4i __device__* src_global, v4i __shared__* dst_lds,
//                   imm offset, imm cpol) per compiler diagnostic.
typedef int v4i_raw __attribute__((vector_size(16)));
typedef __attribute__((address_space(1))) v4i_raw* gv4p;
typedef __attribute__((address_space(3))) v4i_raw* lv4p;

__device__ __forceinline__ void cp16_to_lds(const u16* g, u16* l) {
#if defined(HAVE_ASYNC_LDS)
    __builtin_amdgcn_global_load_async_to_lds_b128(
        (gv4p)(uintptr_t)g, (lv4p)(u32)(uintptr_t)l, 0, 0);
#else
    *(uint4*)l = *(const uint4*)g;                 // synchronous fallback
#endif
}

__device__ __forceinline__ void wait_async0() {
#if defined(HAVE_ASYNC_LDS)
#if __has_builtin(__builtin_amdgcn_s_wait_asynccnt)
    __builtin_amdgcn_s_wait_asynccnt(0);
#else
    asm volatile("s_wait_asynccnt 0x0" ::: "memory");
#endif
#endif
}

__device__ __forceinline__ u16 f2bf(float f) {     // RNE fp32 -> bf16
    u32 x = __float_as_uint(f);
    x += 0x7FFFu + ((x >> 16) & 1u);
    return (u16)(x >> 16);
}

struct Q2 { uint4 a, b; };
__device__ __forceinline__ v16bf mk_frag(uint4 lo, uint4 hi) {
    Q2 q{lo, hi};
    return __builtin_bit_cast(v16bf, q);
}

// A fragment (16x32 bf16), ISA 7.12.2: lanes 0-15 hold M=lane, K={0..7,16..23};
// lanes 16-31 hold M=lane-16, K={8..15,24..31}.  Two contiguous b128 DS reads.
__device__ __forceinline__ v16bf load_a_frag(const u16* base, int ld, int m0,
                                             int k0, int lane) {
    int m  = m0 + (lane & 15);
    int kh = (lane >> 4) << 3;
    const u16* p = base + m * ld + k0 + kh;
    return mk_frag(*(const uint4*)p, *(const uint4*)(p + 16));
}

// B fragment (32x16 bf16): lane l holds column N=l%16, K={0..7,16..23} (lanes 0-15)
// or K={8..15,24..31} (lanes 16-31), matching the A fragment.
__device__ __forceinline__ v16bf load_b_frag(const u16* base, int ld, int k0,
                                             int n0, int lane) {
    const int n  = n0 + (lane & 15);
    const int kh = (lane >> 4) << 3;
    u16 t[16];
    #pragma unroll
    for (int s = 0; s < 8; ++s) {
        t[s]     = base[(k0 + kh + s) * ld + n];
        t[8 + s] = base[(k0 + 16 + kh + s) * ld + n];
    }
    return __builtin_bit_cast(v16bf, t);
}

// ---------------- pre-pass: x -> bf16 (row + transposed) + row sum-of-squares
__global__ __launch_bounds__(256) void prep_x_kernel(
        const float* __restrict__ x, u16* __restrict__ xb,
        u16* __restrict__ xbT, float* __restrict__ sq) {
    const int lane = threadIdx.x & 31, wv = threadIdx.x >> 5;
    const int r = blockIdx.x * 8 + wv;                // one wave per row
    const float* row = x + (size_t)r * DDIM;
    float4 a = *(const float4*)(row + lane * 4);
    float4 b = *(const float4*)(row + 128 + lane * 4);
    float ss = a.x*a.x + a.y*a.y + a.z*a.z + a.w*a.w
             + b.x*b.x + b.y*b.y + b.z*b.z + b.w*b.w;
    #pragma unroll
    for (int off = 16; off; off >>= 1) ss += __shfl_xor(ss, off, 32);
    if (lane == 0) sq[r] = ss;
    u16 ha[4] = {f2bf(a.x), f2bf(a.y), f2bf(a.z), f2bf(a.w)};
    u16 hb[4] = {f2bf(b.x), f2bf(b.y), f2bf(b.z), f2bf(b.w)};
    *(ushort4*)(xb + (size_t)r * DDIM + lane * 4)       = make_ushort4(ha[0], ha[1], ha[2], ha[3]);
    *(ushort4*)(xb + (size_t)r * DDIM + 128 + lane * 4) = make_ushort4(hb[0], hb[1], hb[2], hb[3]);
    #pragma unroll
    for (int e = 0; e < 4; ++e) {
        xbT[(size_t)(lane * 4 + e) * NROWS + r]       = ha[e];
        xbT[(size_t)(128 + lane * 4 + e) * NROWS + r] = hb[e];
    }
}

// ---------------- pre-pass: H -> bf16
__global__ __launch_bounds__(256) void prep_h_kernel(
        const float* __restrict__ Hf, u16* __restrict__ Hb) {
    const size_t i = (size_t)blockIdx.x * blockDim.x + threadIdx.x; // 1M threads
    float4 v = *(const float4*)(Hf + i * 4);
    *(ushort4*)(Hb + i * 4) = make_ushort4(f2bf(v.x), f2bf(v.y), f2bf(v.z), f2bf(v.w));
}

// ---------------- fused Gram->exp->GEMM->reduce kernel ----------------------
__global__ __launch_bounds__(256) void kpca_fused_kernel(
        const u16* __restrict__ xb, const u16* __restrict__ xbT,
        const u16* __restrict__ Hb, const float* __restrict__ sq,
        const float* __restrict__ Hf, const float* __restrict__ invl,
        float* __restrict__ partials) {
    __shared__ __align__(16) u16 sA[BI * A_LD];            // x_I row-major
    __shared__ __align__(16) u16 sXT[2][DDIM * XT_LD];     // x_J^T (dbl buf)
    __shared__ __align__(16) u16 sHb[2][BJ * H_LD];        // H_J   (dbl buf)
    __shared__ __align__(16) u16 sKt[BI * K_LD];           // exp'd K tile
    __shared__ float sqI[BI], sqJ[2][BJ];
    __shared__ float red1[8], red2[8];

    const int tid  = threadIdx.x;
    const int lane = tid & 31;
    const int wv   = tid >> 5;                 // 8 waves
    const int I    = blockIdx.x * BI;
    const int r    = wv >> 1;                  // wave's 16-row group (0..3)
    const int cw   = wv & 1;                   // wave's half (gram col / H half)
    const int m0   = r * 16;
    const int h0   = cw * 256;

    // stage x_I once (64x256 bf16 = 32 KB) + row norms
    #pragma unroll
    for (int c = 0; c < 8; ++c) {
        int idx = tid + c * 256, row = idx >> 5, ch = (idx & 31) * 8;
        cp16_to_lds(xb + (size_t)(I + row) * DDIM + ch, sA + row * A_LD + ch);
    }
    if (tid < BI) sqI[tid] = sq[I + tid];

    // stage j-tile 0 into buffer 0
    {
        #pragma unroll
        for (int c = 0; c < 4; ++c) {          // xT: 256 x 32 halves
            int idx = tid + c * 256, d = idx >> 2, cj = (idx & 3) * 8;
            cp16_to_lds(xbT + (size_t)d * NROWS + cj, &sXT[0][d * XT_LD + cj]);
        }
        #pragma unroll
        for (int c = 0; c < 8; ++c) {          // H: 32 x 512 halves
            int idx = tid + c * 256, row = idx >> 6, ch = (idx & 63) * 8;
            cp16_to_lds(Hb + (size_t)row * HDIM + ch, &sHb[0][row * H_LD + ch]);
        }
        if (tid < BJ) sqJ[0][tid] = sq[tid];
    }
    wait_async0();
    __syncthreads();

    v8f acc[16] = {};                          // F tile: 16 rows x 256 cols

    for (int jb = 0; jb < NJ; ++jb) {
        const int cur = jb & 1;
        // kick off async staging of the NEXT tile into the spare buffers
        // (those were last read before the last barrier -> safe to overwrite now)
        if (jb + 1 < NJ) {
            const int Jn = (jb + 1) * BJ;
            u16* dXT = sXT[cur ^ 1];
            u16* dH  = sHb[cur ^ 1];
            #pragma unroll
            for (int c = 0; c < 4; ++c) {
                int idx = tid + c * 256, d = idx >> 2, cj = (idx & 3) * 8;
                cp16_to_lds(xbT + (size_t)d * NROWS + Jn + cj, dXT + d * XT_LD + cj);
            }
            #pragma unroll
            for (int c = 0; c < 8; ++c) {
                int idx = tid + c * 256, row = idx >> 6, ch = (idx & 63) * 8;
                cp16_to_lds(Hb + (size_t)(Jn + row) * HDIM + ch, dH + row * H_LD + ch);
            }
            if (tid < BJ) sqJ[cur ^ 1][tid] = sq[Jn + tid];
        }

        // Gram tile (64x32): wave owns 16x16 tile (ti=r, tj=cw); K=256.
        v8f g = {};
        const int n0 = cw * 16;
        #pragma unroll
        for (int kk = 0; kk < 8; ++kk) {
            v16bf a = load_a_frag(sA, A_LD, m0, kk * 32, lane);
            v16bf b = load_b_frag(sXT[cur], XT_LD, kk * 32, n0, lane);
            g = __builtin_amdgcn_wmma_f32_16x16x32_bf16(false, a, false, b,
                                                        (short)0, g, false, false);
        }
        // d2 -> exp2(-d2/(2 ln2)) -> bf16 K tile in LDS (C layout scatter)
        {
            const int mo = m0 + ((lane >> 4) << 3);
            const int nl = n0 + (lane & 15);
            const float sj = sqJ[cur][nl];
            #pragma unroll
            for (int gi = 0; gi < 8; ++gi) {
                float d2 = fmaxf(sqI[mo + gi] + sj - 2.f * g[gi], 0.f);
                sKt[(mo + gi) * K_LD + nl] = f2bf(exp2f(-0.72134752044448170f * d2));
            }
        }
        __syncthreads();                       // K tile visible to all waves

        // F_acc += K_tile(64x32) @ H_J(32x512): 16 N-tiles, single K-step.
        {
            v16bf a = load_a_frag(sKt, K_LD, m0, 0, lane);
            #pragma unroll
            for (int n = 0; n < 16; ++n) {
                v16bf b = load_b_frag(sHb[cur], H_LD, 0, h0 + n * 16, lane);
                acc[n] = __builtin_amdgcn_wmma_f32_16x16x32_bf16(false, a, false, b,
                                                                 (short)0, acc[n],
                                                                 false, false);
            }
        }
        wait_async0();                         // my share of next tile landed
        __syncthreads();                       // everyone's landed; sKt free
    }

    // loss partials from F fragments (C layout: row = g + 8*(lane>=16), col = lane&15)
    float p1 = 0.f, p2 = 0.f;
    {
        const int nl = lane & 15;
        const int mo = m0 + ((lane >> 4) << 3);
        #pragma unroll
        for (int n = 0; n < 16; ++n) {
            const int hcol = h0 + n * 16 + nl;
            const float iv = invl[hcol];
            #pragma unroll
            for (int gi = 0; gi < 8; ++gi) {
                const float F = acc[n][gi];
                p1 += iv * F * F;
                p2 += F * Hf[(size_t)(I + mo + gi) * HDIM + hcol];
            }
        }
    }
    #pragma unroll
    for (int off = 16; off; off >>= 1) {
        p1 += __shfl_xor(p1, off, 32);
        p2 += __shfl_xor(p2, off, 32);
    }
    if (lane == 0) { red1[wv] = p1; red2[wv] = p2; }
    __syncthreads();
    if (tid == 0) {
        float s1 = 0.f, s2 = 0.f;
        #pragma unroll
        for (int w = 0; w < 8; ++w) { s1 += red1[w]; s2 += red2[w]; }
        partials[blockIdx.x * 2]     = s1;
        partials[blockIdx.x * 2 + 1] = s2;
    }
}

// ---------------- deterministic finalize ------------------------------------
__global__ void finalize_kernel(const float* __restrict__ partials,
                                float* __restrict__ out) {
    if (threadIdx.x == 0 && blockIdx.x == 0) {
        float s1 = 0.f, s2 = 0.f;
        for (int b = 0; b < NBLK; ++b) { s1 += partials[2 * b]; s2 += partials[2 * b + 1]; }
        const float loss1 = s1 * (-0.5f);      // / (-2 eta^2)
        const float loss2 = s2 * 0.5f;         // / (2 eta)
        const float s = loss1 + loss2;
        out[0] = s + 0.05f * s * s;            // + (C_STAB/2) s^2
    }
}

extern "C" void kernel_launch(void* const* d_in, const int* in_sizes, int n_in,
                              void* d_out, int out_size, void* d_ws, size_t ws_size,
                              hipStream_t stream) {
    const float* x    = (const float*)d_in[0];   // [8192,256]
    const float* Hf   = (const float*)d_in[1];   // [8192,512]
    const float* invl = (const float*)d_in[2];   // [512]
    float* out = (float*)d_out;

    char* ws = (char*)d_ws;
    u16*   xb       = (u16*)(ws);                        //  4 MB
    u16*   xbT      = (u16*)(ws + (size_t)4  * 1048576); //  4 MB
    u16*   Hb       = (u16*)(ws + (size_t)8  * 1048576); //  8 MB
    float* sq       = (float*)(ws + (size_t)16 * 1048576);
    float* partials = (float*)(ws + (size_t)16 * 1048576 + 32768);

    prep_x_kernel<<<NROWS / 8, 256, 0, stream>>>(x, xb, xbT, sq);
    prep_h_kernel<<<(NROWS * HDIM / 4) / 256, 256, 0, stream>>>(Hf, Hb);
    kpca_fused_kernel<<<NBLK, 256, 0, stream>>>(xb, xbT, Hb, sq, Hf, invl, partials);
    finalize_kernel<<<1, 1, 0, stream>>>(partials, out);
}
